// Model_86964497809601
// MI455X (gfx1250) — matrix operand order repaired and checked by reading, not compiled, not run on hardware
//
#include <hip/hip_runtime.h>
#include <hip/hip_bf16.h>

// ---------------------------------------------------------------------------
// GIN (3-layer) + projection head for MI455X (gfx1250, wave32, WMMA).
//  - Edge aggregation: CSR gather, one wave per node (coalesced 512B row reads)
//  - GEMMs: V_WMMA_F32_16X16X32_F16, B^T staged in LDS (async-to-LDS if avail)
// ---------------------------------------------------------------------------

typedef _Float16 h4  __attribute__((ext_vector_type(4)));
typedef _Float16 h8  __attribute__((ext_vector_type(8)));
typedef _Float16 h16 __attribute__((ext_vector_type(16)));
typedef float    f8  __attribute__((ext_vector_type(8)));
typedef int      vs4i __attribute__((vector_size(16)));   // matches builtin param

#define N_NODES 100000
#define N_EDGES 1600000
#define IN_DIM  200
#define IN_PAD  224      // 200 padded to multiple of 32
#define HID     128
#define N_GRAPHS 64
#define CAT_DIM 384      // HID * 3

#if defined(__has_builtin)
#if __has_builtin(__builtin_amdgcn_global_load_async_to_lds_b128) && \
    __has_builtin(__builtin_amdgcn_s_wait_asynccnt)
#define HAVE_ASYNC_LDS 1
#endif
#endif

// ---------------------------------------------------------------------------
// small helpers
// ---------------------------------------------------------------------------
__global__ void zero_f32_kernel(float* __restrict__ p, int n) {
    int t = blockIdx.x * 256 + threadIdx.x;
    if (t < n) p[t] = 0.0f;
}
__global__ void zero_i32_kernel(int* __restrict__ p, int n) {
    int t = blockIdx.x * 256 + threadIdx.x;
    if (t < n) p[t] = 0;
}
__global__ void copy_i32_kernel(int* __restrict__ dst, const int* __restrict__ src, int n) {
    int t = blockIdx.x * 256 + threadIdx.x;
    if (t < n) dst[t] = src[t];
}

// ---------------------------------------------------------------------------
// CSR construction (per-launch)
// ---------------------------------------------------------------------------
__global__ void count_deg_kernel(const int* __restrict__ ei, int E, int* __restrict__ deg) {
    int e = blockIdx.x * 256 + threadIdx.x;
    if (e < E) atomicAdd(&deg[ei[e]], 1);
}

// per-block exclusive scan of 256 ints + block total
__global__ void scan_block_kernel(const int* __restrict__ deg, int* __restrict__ excl,
                                  int* __restrict__ bsum, int n) {
    __shared__ int s[256];
    int gid = blockIdx.x * 256 + threadIdx.x;
    int v = (gid < n) ? deg[gid] : 0;
    s[threadIdx.x] = v;
    __syncthreads();
    for (int off = 1; off < 256; off <<= 1) {
        int t = (threadIdx.x >= off) ? s[threadIdx.x - off] : 0;
        __syncthreads();
        s[threadIdx.x] += t;
        __syncthreads();
    }
    if (gid < n) excl[gid] = s[threadIdx.x] - v;
    if (threadIdx.x == 255) bsum[blockIdx.x] = s[255];
}

__global__ void scan_bsum_kernel(int* __restrict__ bsum, int nb, int* __restrict__ total_out) {
    if (blockIdx.x == 0 && threadIdx.x == 0) {
        int run = 0;
        for (int b = 0; b < nb; ++b) { int t = bsum[b]; bsum[b] = run; run += t; }
        total_out[0] = run;   // rowptr[N] = E
    }
}

__global__ void scan_add_kernel(int* __restrict__ excl, const int* __restrict__ bsum, int n) {
    int t = blockIdx.x * 256 + threadIdx.x;
    if (t < n) excl[t] += bsum[t >> 8];
}

__global__ void fill_csr_kernel(const int* __restrict__ ei, int E,
                                int* __restrict__ cursor, int* __restrict__ colidx) {
    int e = blockIdx.x * 256 + threadIdx.x;
    if (e >= E) return;
    int r = ei[e];
    int pos = atomicAdd(&cursor[r], 1);
    colidx[pos] = ei[E + e];
}

// ---------------------------------------------------------------------------
// CSR-gather aggregation: pooled = (1+eps)*h[n] + sum_{j in N(n)} h[col[j]]
// One wave per node; lane = 4-channel chunk; neighbor rows read coalesced.
// Output written directly as padded f16 (GEMM A operand).
// ---------------------------------------------------------------------------
__global__ void pad224_zero_kernel(_Float16* __restrict__ A16, int n /* M*24 */) {
    int t = blockIdx.x * 256 + threadIdx.x;
    if (t >= n) return;
    int m = t / 24;
    int k = 200 + (t - m * 24);
    A16[(size_t)m * IN_PAD + k] = (_Float16)0.0f;
}

__global__ __launch_bounds__(256)
void agg200_kernel(const float* __restrict__ x, const int* __restrict__ rowptr,
                   const int* __restrict__ colidx, const float* __restrict__ eps,
                   _Float16* __restrict__ A16, int M) {
    const int wave = threadIdx.x >> 5;
    const int lane = threadIdx.x & 31;
    const int n = blockIdx.x * 8 + wave;
    if (n >= M) return;
    const float se = 1.0f + eps[0];
    const int c0 = lane * 4;
    const int c1 = 128 + lane * 4;          // second chunk, valid for lane<18
    const bool has2 = (lane < 18);
    const float* xr = x + (size_t)n * IN_DIM;
    float4 a0 = *(const float4*)(xr + c0);
    a0.x *= se; a0.y *= se; a0.z *= se; a0.w *= se;
    float4 a1 = make_float4(0.f, 0.f, 0.f, 0.f);
    if (has2) {
        a1 = *(const float4*)(xr + c1);
        a1.x *= se; a1.y *= se; a1.z *= se; a1.w *= se;
    }
    const int s = rowptr[n], e = rowptr[n + 1];
    for (int j = s; j < e; ++j) {
        const float* hr = x + (size_t)colidx[j] * IN_DIM;
        float4 v0 = *(const float4*)(hr + c0);
        a0.x += v0.x; a0.y += v0.y; a0.z += v0.z; a0.w += v0.w;
        if (has2) {
            float4 v1 = *(const float4*)(hr + c1);
            a1.x += v1.x; a1.y += v1.y; a1.z += v1.z; a1.w += v1.w;
        }
    }
    _Float16* Ar = A16 + (size_t)n * IN_PAD;
    h4 o0; o0[0] = (_Float16)a0.x; o0[1] = (_Float16)a0.y; o0[2] = (_Float16)a0.z; o0[3] = (_Float16)a0.w;
    *(h4*)(Ar + c0) = o0;
    if (has2) {
        h4 o1; o1[0] = (_Float16)a1.x; o1[1] = (_Float16)a1.y; o1[2] = (_Float16)a1.z; o1[3] = (_Float16)a1.w;
        *(h4*)(Ar + c1) = o1;
    }
}

__global__ __launch_bounds__(256)
void agg128_kernel(const float* __restrict__ h, const int* __restrict__ rowptr,
                   const int* __restrict__ colidx, const float* __restrict__ eps, int l,
                   _Float16* __restrict__ A16, int M) {
    const int wave = threadIdx.x >> 5;
    const int lane = threadIdx.x & 31;
    const int n = blockIdx.x * 8 + wave;
    if (n >= M) return;
    const float se = 1.0f + eps[l];
    const int c = lane * 4;
    float4 a = *(const float4*)(h + (size_t)n * HID + c);
    a.x *= se; a.y *= se; a.z *= se; a.w *= se;
    const int s = rowptr[n], e = rowptr[n + 1];
    for (int j = s; j < e; ++j) {
        float4 v = *(const float4*)(h + (size_t)colidx[j] * HID + c);
        a.x += v.x; a.y += v.y; a.z += v.z; a.w += v.w;
    }
    h4 o; o[0] = (_Float16)a.x; o[1] = (_Float16)a.y; o[2] = (_Float16)a.z; o[3] = (_Float16)a.w;
    *(h4*)(A16 + (size_t)n * HID + c) = o;
}

// ---------------------------------------------------------------------------
// weight / misc conversions
// ---------------------------------------------------------------------------
__global__ void cvt_pad_kernel(_Float16* __restrict__ dst, const float* __restrict__ src,
                               int K, int Kpad, int n) {
    int t = blockIdx.x * 256 + threadIdx.x;
    if (t >= n) return;
    int m = t / Kpad;
    int k = t - m * Kpad;
    dst[t] = (k < K) ? (_Float16)src[(size_t)m * K + k] : (_Float16)0.0f;
}

__global__ void cvt_transpose_kernel(_Float16* __restrict__ dst, const float* __restrict__ src,
                                     int K, int Kpad) {
    int t = blockIdx.x * 256 + threadIdx.x;
    int n = HID * Kpad;
    if (t >= n) return;
    int ncol = t / Kpad;
    int k    = t - ncol * Kpad;
    dst[t] = (k < K) ? (_Float16)src[(size_t)k * HID + ncol] : (_Float16)0.0f;
}

__global__ void relu_cvt_kernel(_Float16* __restrict__ dst, const float* __restrict__ src, int n) {
    int t = blockIdx.x * 256 + threadIdx.x;
    if (t < n) dst[t] = (_Float16)fmaxf(src[t], 0.0f);
}

// ---------------------------------------------------------------------------
// WMMA GEMM with B^T staged in LDS (padded stride, conflict-free).
// Fragment layouts per CDNA5 ISA 7.12.2.
// ---------------------------------------------------------------------------
template <int KPAD>
__global__ __launch_bounds__(256)
void gemm_wmma_lds_kernel(const _Float16* __restrict__ A, const _Float16* __restrict__ BT,
                          const float* __restrict__ bias, float* __restrict__ Y, int M) {
    constexpr int SBS = KPAD + 8;                 // padded row stride (halves)
    __shared__ _Float16 sB[HID * SBS];

    // stage B^T [128 x KPAD] into LDS, 16B chunks per thread
    constexpr int CHUNKS = HID * (KPAD / 8);
    for (int i = threadIdx.x; i < CHUNKS; i += 256) {
        int nrow = i / (KPAD / 8);
        int kc   = i - nrow * (KPAD / 8);
#ifdef HAVE_ASYNC_LDS
        __builtin_amdgcn_global_load_async_to_lds_b128(
            (__attribute__((address_space(1))) vs4i*)(BT + (size_t)nrow * KPAD + kc * 8),
            (__attribute__((address_space(3))) vs4i*)(sB + nrow * SBS + kc * 8),
            0, 0);
#else
        *(h8*)(sB + nrow * SBS + kc * 8) = *(const h8*)(BT + (size_t)nrow * KPAD + kc * 8);
#endif
    }
#ifdef HAVE_ASYNC_LDS
    __builtin_amdgcn_s_wait_asynccnt(0);
#endif
    __syncthreads();

    const int wave = threadIdx.x >> 5;
    const int lane = threadIdx.x & 31;
    const int m0 = (blockIdx.x * 8 + wave) * 16;
    if (m0 >= M) return;                     // wave-uniform (after the barrier)

    const int mr  = lane & 15;
    const int hiA = (lane >> 4) * 8;
    const int hiB = (lane >> 4) * 8;
    const _Float16* Arow = A + (size_t)(m0 + mr) * KPAD;

    f8 acc[8] = {};

    for (int k0 = 0; k0 < KPAD; k0 += 32) {
        h8 a_lo = *(const h8*)(Arow + k0 + hiA);
        h8 a_hi = *(const h8*)(Arow + k0 + 16 + hiA);
        h16 a = __builtin_shufflevector(a_lo, a_hi,
                 0,1,2,3,4,5,6,7,8,9,10,11,12,13,14,15);
        #pragma unroll
        for (int nt = 0; nt < 8; ++nt) {
            const _Float16* Bp = sB + (nt * 16 + mr) * SBS + k0 + hiB;
            h8 b_lo = *(const h8*)(Bp);
            h8 b_hi = *(const h8*)(Bp + 16);
            h16 b = __builtin_shufflevector(b_lo, b_hi,
                     0,1,2,3,4,5,6,7,8,9,10,11,12,13,14,15);
            acc[nt] = __builtin_amdgcn_wmma_f32_16x16x32_f16(
                false, a, false, b, (short)0, acc[nt], false, false);
        }
    }

    const int rbase = (lane >> 4) * 8;
    #pragma unroll
    for (int nt = 0; nt < 8; ++nt) {
        const int col = nt * 16 + mr;
        const float bv = bias[col];
        #pragma unroll
        for (int r = 0; r < 8; ++r) {
            Y[(size_t)(m0 + rbase + r) * HID + col] = acc[nt][r] + bv;
        }
    }
}

// direct-VMEM variant (runtime Kpad) for the 384-wide head GEMM
__global__ __launch_bounds__(256)
void gemm_wmma_direct_kernel(const _Float16* __restrict__ A, const _Float16* __restrict__ BT,
                             const float* __restrict__ bias, float* __restrict__ Y,
                             int M, int Kpad) {
    const int wave = threadIdx.x >> 5;
    const int lane = threadIdx.x & 31;
    const int m0 = (blockIdx.x * 8 + wave) * 16;
    if (m0 >= M) return;
    const int mr  = lane & 15;
    const int hiA = (lane >> 4) * 8;
    const int hiB = (lane >> 4) * 8;
    const _Float16* Arow = A + (size_t)(m0 + mr) * Kpad;
    f8 acc[8] = {};
    for (int k0 = 0; k0 < Kpad; k0 += 32) {
        h8 a_lo = *(const h8*)(Arow + k0 + hiA);
        h8 a_hi = *(const h8*)(Arow + k0 + 16 + hiA);
        h16 a = __builtin_shufflevector(a_lo, a_hi,
                 0,1,2,3,4,5,6,7,8,9,10,11,12,13,14,15);
        #pragma unroll
        for (int nt = 0; nt < 8; ++nt) {
            const _Float16* Bp = BT + (size_t)(nt * 16 + mr) * Kpad + k0 + hiB;
            h8 b_lo = *(const h8*)(Bp);
            h8 b_hi = *(const h8*)(Bp + 16);
            h16 b = __builtin_shufflevector(b_lo, b_hi,
                     0,1,2,3,4,5,6,7,8,9,10,11,12,13,14,15);
            acc[nt] = __builtin_amdgcn_wmma_f32_16x16x32_f16(
                false, a, false, b, (short)0, acc[nt], false, false);
        }
    }
    const int rbase = (lane >> 4) * 8;
    #pragma unroll
    for (int nt = 0; nt < 8; ++nt) {
        const int col = nt * 16 + mr;
        const float bv = bias[col];
        #pragma unroll
        for (int r = 0; r < 8; ++r) {
            Y[(size_t)(m0 + rbase + r) * HID + col] = acc[nt][r] + bv;
        }
    }
}

// ---------------------------------------------------------------------------
// BatchNorm (train-mode, biased variance over rows)
// ---------------------------------------------------------------------------
__global__ void bn_stats_kernel(const float* __restrict__ Y, int M,
                                float* __restrict__ sum, float* __restrict__ sumsq) {
    __shared__ float s1[256];
    __shared__ float s2[256];
    const int c    = threadIdx.x & 127;
    const int half = threadIdx.x >> 7;
    float a = 0.0f, b = 0.0f;
    for (int r = blockIdx.x * 2 + half; r < M; r += gridDim.x * 2) {
        float v = Y[(size_t)r * HID + c];
        a += v;
        b += v * v;
    }
    s1[threadIdx.x] = a;
    s2[threadIdx.x] = b;
    __syncthreads();
    if (half == 0) {
        a += s1[threadIdx.x + 128];
        b += s2[threadIdx.x + 128];
        atomicAdd(&sum[c], a);
        atomicAdd(&sumsq[c], b);
    }
}

__global__ void bn_finalize_kernel(const float* __restrict__ sum, const float* __restrict__ sumsq,
                                   const float* __restrict__ g, const float* __restrict__ be,
                                   int M, float* __restrict__ scale, float* __restrict__ shift) {
    int c = threadIdx.x;
    if (c >= HID) return;
    float inv_m = 1.0f / (float)M;
    float mean = sum[c] * inv_m;
    float var  = sumsq[c] * inv_m - mean * mean;
    float rstd = rsqrtf(var + 1e-5f);
    float sc = g[c] * rstd;
    scale[c] = sc;
    shift[c] = be[c] - mean * sc;
}

__global__ void bn_apply_f16_kernel(const float* __restrict__ Y, const float* __restrict__ scale,
                                    const float* __restrict__ shift, int n,
                                    _Float16* __restrict__ out16) {
    int t = blockIdx.x * 256 + threadIdx.x;
    if (t >= n) return;
    int c = t & 127;
    out16[t] = (_Float16)fmaxf(Y[t] * scale[c] + shift[c], 0.0f);
}

__global__ void bn_apply_f32_kernel(const float* __restrict__ Y, const float* __restrict__ scale,
                                    const float* __restrict__ shift, int n,
                                    float* __restrict__ out) {
    int t = blockIdx.x * 256 + threadIdx.x;
    if (t >= n) return;
    int c = t & 127;
    out[t] = fmaxf(Y[t] * scale[c] + shift[c], 0.0f);
}

// global_add_pool: pg[batch[node]][layer*128 + c] += h[node][c]
__global__ void graph_pool_kernel(const float* __restrict__ h, const int* __restrict__ batch,
                                  float* __restrict__ pg, int layer, int M) {
    int t = blockIdx.x * 256 + threadIdx.x;
    int total = M * 32;
    if (t >= total) return;
    int node = t >> 5;
    int c = (t & 31) * 4;
    int g = batch[node];
    const float4 v = *(const float4*)(h + (size_t)node * HID + c);
    float* p = pg + (size_t)g * CAT_DIM + layer * HID + c;
    atomicAdd(p + 0, v.x);
    atomicAdd(p + 1, v.y);
    atomicAdd(p + 2, v.z);
    atomicAdd(p + 3, v.w);
}

// ---------------------------------------------------------------------------
// host-side orchestration
// ---------------------------------------------------------------------------
static inline dim3 gsz(long long n) { return dim3((unsigned)((n + 255) / 256)); }

extern "C" void kernel_launch(void* const* d_in, const int* in_sizes, int n_in,
                              void* d_out, int out_size, void* d_ws, size_t ws_size,
                              hipStream_t stream) {
    const float* x      = (const float*)d_in[0];
    const int*   ei     = (const int*)  d_in[1];
    const int*   batch  = (const int*)  d_in[2];
    const float* eps    = (const float*)d_in[3];
    const float* w1_0   = (const float*)d_in[4];
    const float* b1_0   = (const float*)d_in[5];
    const float* g1_0   = (const float*)d_in[6];
    const float* be1_0  = (const float*)d_in[7];
    const float* w2_0   = (const float*)d_in[8];
    const float* b2_0   = (const float*)d_in[9];
    const float* go_0   = (const float*)d_in[10];
    const float* bo_0   = (const float*)d_in[11];
    const float* w1_r   = (const float*)d_in[12];
    const float* b1_r   = (const float*)d_in[13];
    const float* g1_r   = (const float*)d_in[14];
    const float* be1_r  = (const float*)d_in[15];
    const float* w2_r   = (const float*)d_in[16];
    const float* b2_r   = (const float*)d_in[17];
    const float* go_r   = (const float*)d_in[18];
    const float* bo_r   = (const float*)d_in[19];
    const float* pw1    = (const float*)d_in[20];
    const float* pb1    = (const float*)d_in[21];
    const float* pw2    = (const float*)d_in[22];
    const float* pb2    = (const float*)d_in[23];
    float* out = (float*)d_out;

    // ---- workspace layout (256B aligned) ----
    char* ws = (char*)d_ws;
    const size_t off_h    = 0;                                          // f32 [100000 x 128]
    const size_t off_A16  = off_h    + (size_t)N_NODES * HID * 4;       // f16 [100000 x 224] (reused as Z16)
    const size_t off_Y    = off_A16  + (size_t)N_NODES * IN_PAD * 2;    // f32 [100000 x 128]
    const size_t off_W16  = off_Y    + (size_t)N_NODES * HID * 4;       // f16 [128 x 384] max
    const size_t off_stat = off_W16  + 131072;
    const size_t off_pg   = off_stat + 2048;                            // f32 [64 x 384]
    const size_t off_cat  = off_pg   + (size_t)N_GRAPHS * CAT_DIM * 4;  // f16 [64 x 384]
    const size_t off_t16  = off_cat  + (size_t)N_GRAPHS * CAT_DIM * 2;  // f16 [64 x 128]
    const size_t off_rp   = off_t16  + 16640;                           // int [100001]
    const size_t off_cur  = off_rp   + 400384;                          // int [100000] (deg, then cursor)
    const size_t off_col  = off_cur  + 400384;                          // int [1600000]
    const size_t off_bs   = off_col  + (size_t)N_EDGES * 4;             // int [391]

    float*     h     = (float*)(ws + off_h);
    _Float16*  A16   = (_Float16*)(ws + off_A16);       // also Z16
    float*     Y     = (float*)(ws + off_Y);
    _Float16*  W16   = (_Float16*)(ws + off_W16);
    float*     sum   = (float*)(ws + off_stat);
    float*     sumsq = sum + HID;
    float*     scale = sum + 2 * HID;
    float*     shift = sum + 3 * HID;
    float*     pg    = (float*)(ws + off_pg);
    _Float16*  cat16 = (_Float16*)(ws + off_cat);
    _Float16*  t16   = (_Float16*)(ws + off_t16);
    int*       rowptr = (int*)(ws + off_rp);
    int*       cursor = (int*)(ws + off_cur);
    int*       colidx = (int*)(ws + off_col);
    int*       bsum   = (int*)(ws + off_bs);

    const int gemm_blocks = (N_NODES / 16 + 7) / 8;     // 782
    const int agg_blocks  = (N_NODES + 7) / 8;          // 12500
    const int nb = (N_NODES + 255) / 256;               // 391

    // ---- CSR build (amortized over all 3 layers) ----
    zero_i32_kernel<<<gsz(N_NODES), 256, 0, stream>>>(cursor, N_NODES);               // cursor = deg
    count_deg_kernel<<<gsz(N_EDGES), 256, 0, stream>>>(ei, N_EDGES, cursor);
    scan_block_kernel<<<nb, 256, 0, stream>>>(cursor, rowptr, bsum, N_NODES);
    scan_bsum_kernel<<<1, 32, 0, stream>>>(bsum, nb, rowptr + N_NODES);
    scan_add_kernel<<<gsz(N_NODES), 256, 0, stream>>>(rowptr, bsum, N_NODES);
    copy_i32_kernel<<<gsz(N_NODES), 256, 0, stream>>>(cursor, rowptr, N_NODES);
    fill_csr_kernel<<<gsz(N_EDGES), 256, 0, stream>>>(ei, N_EDGES, cursor, colidx);

    zero_f32_kernel<<<gsz(N_GRAPHS * CAT_DIM), 256, 0, stream>>>(pg, N_GRAPHS * CAT_DIM);

    // ================= layer 0 (IN_DIM=200) =================
    pad224_zero_kernel<<<gsz((long long)N_NODES * 24), 256, 0, stream>>>(A16, N_NODES * 24);
    agg200_kernel<<<agg_blocks, 256, 0, stream>>>(x, rowptr, colidx, eps, A16, N_NODES);
    cvt_transpose_kernel<<<gsz(HID * IN_PAD), 256, 0, stream>>>(W16, w1_0, IN_DIM, IN_PAD);
    gemm_wmma_lds_kernel<IN_PAD><<<gemm_blocks, 256, 0, stream>>>(A16, W16, b1_0, Y, N_NODES);

    zero_f32_kernel<<<1, 256, 0, stream>>>(sum, 2 * HID);
    bn_stats_kernel<<<512, 256, 0, stream>>>(Y, N_NODES, sum, sumsq);
    bn_finalize_kernel<<<1, 128, 0, stream>>>(sum, sumsq, g1_0, be1_0, N_NODES, scale, shift);
    bn_apply_f16_kernel<<<gsz((long long)N_NODES * HID), 256, 0, stream>>>(Y, scale, shift, N_NODES * HID, A16);

    cvt_transpose_kernel<<<gsz(HID * HID), 256, 0, stream>>>(W16, w2_0, HID, HID);
    gemm_wmma_lds_kernel<HID><<<gemm_blocks, 256, 0, stream>>>(A16, W16, b2_0, Y, N_NODES);

    zero_f32_kernel<<<1, 256, 0, stream>>>(sum, 2 * HID);
    bn_stats_kernel<<<512, 256, 0, stream>>>(Y, N_NODES, sum, sumsq);
    bn_finalize_kernel<<<1, 128, 0, stream>>>(sum, sumsq, go_0, bo_0, N_NODES, scale, shift);
    bn_apply_f32_kernel<<<gsz((long long)N_NODES * HID), 256, 0, stream>>>(Y, scale, shift, N_NODES * HID, h);

    graph_pool_kernel<<<gsz((long long)N_NODES * 32), 256, 0, stream>>>(h, batch, pg, 0, N_NODES);

    // ================= layers 1..2 (HID=128) =================
    for (int r = 0; r < 2; ++r) {
        agg128_kernel<<<agg_blocks, 256, 0, stream>>>(h, rowptr, colidx, eps, r + 1, A16, N_NODES);

        cvt_transpose_kernel<<<gsz(HID * HID), 256, 0, stream>>>(W16, w1_r + (size_t)r * HID * HID, HID, HID);
        gemm_wmma_lds_kernel<HID><<<gemm_blocks, 256, 0, stream>>>(A16, W16, b1_r + (size_t)r * HID, Y, N_NODES);

        zero_f32_kernel<<<1, 256, 0, stream>>>(sum, 2 * HID);
        bn_stats_kernel<<<512, 256, 0, stream>>>(Y, N_NODES, sum, sumsq);
        bn_finalize_kernel<<<1, 128, 0, stream>>>(sum, sumsq, g1_r + (size_t)r * HID, be1_r + (size_t)r * HID, N_NODES, scale, shift);
        bn_apply_f16_kernel<<<gsz((long long)N_NODES * HID), 256, 0, stream>>>(Y, scale, shift, N_NODES * HID, A16);

        cvt_transpose_kernel<<<gsz(HID * HID), 256, 0, stream>>>(W16, w2_r + (size_t)r * HID * HID, HID, HID);
        gemm_wmma_lds_kernel<HID><<<gemm_blocks, 256, 0, stream>>>(A16, W16, b2_r + (size_t)r * HID, Y, N_NODES);

        zero_f32_kernel<<<1, 256, 0, stream>>>(sum, 2 * HID);
        bn_stats_kernel<<<512, 256, 0, stream>>>(Y, N_NODES, sum, sumsq);
        bn_finalize_kernel<<<1, 128, 0, stream>>>(sum, sumsq, go_r + (size_t)r * HID, bo_r + (size_t)r * HID, N_NODES, scale, shift);
        bn_apply_f32_kernel<<<gsz((long long)N_NODES * HID), 256, 0, stream>>>(Y, scale, shift, N_NODES * HID, h);

        graph_pool_kernel<<<gsz((long long)N_NODES * 32), 256, 0, stream>>>(h, batch, pg, r + 1, N_NODES);
    }

    // ================= projection head =================
    cvt_pad_kernel<<<gsz(N_GRAPHS * CAT_DIM), 256, 0, stream>>>(cat16, pg, CAT_DIM, CAT_DIM, N_GRAPHS * CAT_DIM);
    cvt_transpose_kernel<<<gsz(HID * CAT_DIM), 256, 0, stream>>>(W16, pw1, CAT_DIM, CAT_DIM);
    gemm_wmma_direct_kernel<<<1, 256, 0, stream>>>(cat16, W16, pb1, Y, N_GRAPHS, CAT_DIM);
    relu_cvt_kernel<<<gsz(N_GRAPHS * HID), 256, 0, stream>>>(t16, Y, N_GRAPHS * HID);
    cvt_transpose_kernel<<<gsz(HID * HID), 256, 0, stream>>>(W16, pw2, HID, HID);
    gemm_wmma_lds_kernel<HID><<<1, 256, 0, stream>>>(t16, W16, pb2, out, N_GRAPHS);

    (void)in_sizes; (void)n_in; (void)out_size; (void)ws_size;
}
